// GINClassification_21861383536763
// MI455X (gfx1250) — matrix operand order repaired and checked by reading, not compiled, not run on hardware
//
#include <hip/hip_runtime.h>
#include <hip/hip_bf16.h>
#include <stdint.h>

// ---------------------------------------------------------------------------
// GIN (3x GINConv + mean-pool + linear classifier) for MI455X / gfx1250.
//  * scatter-add aggregation: 32 lanes/edge, float4 gather + f32 atomics
//  * Linear+bias+BN-stats fused GEMM using v_wmma_f32_16x16x32_f16 (wave32)
//  * BN finalize + fused BN*ReLU elementwise pass
//  * per-graph mean pool via atomics, tiny classifier kernel
// ---------------------------------------------------------------------------

typedef __attribute__((ext_vector_type(16))) _Float16 v16h;
typedef __attribute__((ext_vector_type(8)))  float    v8f;

#define FDIM   128      // IN == H == OUT == 128
#define NGRAPH 64
#define NCLS   10
#define BN_EPS 1e-5f

// ------------------------- GIN aggregation (scatter) ------------------------
// H must be pre-initialized to X (residual / eps=0 term). 32 threads per edge,
// each thread moves 4 contiguous floats -> fully coalesced 512B per wavefront.
__global__ void k_scatter_add(const float* __restrict__ X,
                              const long long* __restrict__ src,
                              const long long* __restrict__ dst,
                              float* __restrict__ H, long long nedges)
{
    long long gid = (long long)blockIdx.x * blockDim.x + threadIdx.x;
    long long e   = gid >> 5;
    if (e >= nedges) return;
    int f4 = (int)(gid & 31) << 2;
    long long s = src[e], d = dst[e];
    float4 v = *(const float4*)(X + s * FDIM + f4);
    float* out = H + d * FDIM + f4;
    atomicAdd(out + 0, v.x);
    atomicAdd(out + 1, v.y);
    atomicAdd(out + 2, v.z);
    atomicAdd(out + 3, v.w);
}

// --------------- GEMM (N x 128) @ (128 x 128) + bias + BN stats -------------
// Block = 256 threads = 8 wave32's. Wave w computes rows [.. +16) x all 128
// cols as 8 accumulator tiles of v_wmma_f32_16x16x32_f16 (K-loop of 4).
// Weights are transposed to f16 in LDS so each lane's B fragment is one
// contiguous 32B LDS load. Epilogue: +bias, store T, accumulate per-column
// sum / sum-of-squares for training-mode BatchNorm.
__global__ void __launch_bounds__(256)
k_gemm_bias_stats(const float* __restrict__ A,     // [nrows,128] f32
                  const float* __restrict__ W,     // [128,128]  f32 (k-major)
                  const float* __restrict__ bias,  // [128]
                  float* __restrict__ T,           // [nrows,128] out
                  float* __restrict__ colsum,      // [128] (pre-zeroed)
                  float* __restrict__ colsq,       // [128] (pre-zeroed)
                  int nrows)
{
    __shared__ __attribute__((aligned(32))) _Float16 Wt[FDIM * FDIM]; // 32 KB

    const int tid = threadIdx.x;
    // cooperative load + transpose + f32->f16 convert of the weights
    for (int i = tid; i < FDIM * FDIM; i += 256) {
        int k = i >> 7, n = i & (FDIM - 1);
        Wt[n * FDIM + k] = (_Float16)W[i];
    }
    __syncthreads();

    const int wave = tid >> 5;
    const int lane = tid & 31;
    const int half = lane >> 4;   // 0: lanes 0-15, 1: lanes 16-31
    const int lr   = lane & 15;

    const int row0 = (blockIdx.x * 8 + wave) * 16;
    if (row0 >= nrows) return;    // wave-uniform exit (after barrier)

    int arow = row0 + lr;
    if (arow >= nrows) arow = nrows - 1;          // clamp (EXEC stays full)
    const float* __restrict__ Ar = A + (size_t)arow * FDIM;
    __builtin_prefetch(Ar, 0, 0);                 // global_prefetch_b8

    v8f acc[8] = {};

    #pragma unroll
    for (int k0 = 0; k0 < FDIM; k0 += 32) {
        // A fragment, 16-bit A 16x32 layout (contiguous, matches B):
        //   lane<16 : elems 0..15 -> K=k0..k0+15
        //   lane>=16: elems 0..15 -> K=k0+16..k0+31
        const int kb0 = k0 + half * 16;
        const int kb1 = kb0 + 8;
        float4 a0 = *(const float4*)(Ar + kb0);
        float4 a1 = *(const float4*)(Ar + kb0 + 4);
        float4 a2 = *(const float4*)(Ar + kb1);
        float4 a3 = *(const float4*)(Ar + kb1 + 4);
        v16h a;
        a[0]  = (_Float16)a0.x; a[1]  = (_Float16)a0.y;
        a[2]  = (_Float16)a0.z; a[3]  = (_Float16)a0.w;
        a[4]  = (_Float16)a1.x; a[5]  = (_Float16)a1.y;
        a[6]  = (_Float16)a1.z; a[7]  = (_Float16)a1.w;
        a[8]  = (_Float16)a2.x; a[9]  = (_Float16)a2.y;
        a[10] = (_Float16)a2.z; a[11] = (_Float16)a2.w;
        a[12] = (_Float16)a3.x; a[13] = (_Float16)a3.y;
        a[14] = (_Float16)a3.z; a[15] = (_Float16)a3.w;

        // B fragment, contiguous k order (matches A): lane<16 col=lr K=k0..k0+15,
        // lane>=16 col=lr K=k0+16..k0+31 -> one contiguous 32B LDS read.
        const int kb = k0 + half * 16;
        #pragma unroll
        for (int nt = 0; nt < 8; ++nt) {
            const int col = nt * 16 + lr;
            v16h b = *(const v16h*)(&Wt[col * FDIM + kb]);
            acc[nt] = __builtin_amdgcn_wmma_f32_16x16x32_f16(
                /*neg_a=*/false, a, /*neg_b=*/false, b,
                /*c_mod=*/(short)0, acc[nt],
                /*reuse_a=*/false, /*reuse_b=*/false);
        }
    }

    // Epilogue: C layout — VGPR r holds row (row0 + half*8 + r), col = nt*16+lr
    #pragma unroll
    for (int nt = 0; nt < 8; ++nt) {
        const int col = nt * 16 + lr;
        const float bc = bias[col];
        float s = 0.0f, sq = 0.0f;
        #pragma unroll
        for (int r = 0; r < 8; ++r) {
            const int row = row0 + half * 8 + r;
            const float v = acc[nt][r] + bc;
            if (row < nrows) {
                T[(size_t)row * FDIM + col] = v;
                s  += v;
                sq += v * v;
            }
        }
        atomicAdd(&colsum[col], s);
        atomicAdd(&colsq[col],  sq);
    }
}

// ----------------------------- BN finalize ----------------------------------
__global__ void k_bn_finalize(const float* __restrict__ colsum,
                              const float* __restrict__ colsq,
                              const float* __restrict__ gamma,
                              const float* __restrict__ beta,
                              float* __restrict__ scale,
                              float* __restrict__ shift, float inv_n)
{
    int f = threadIdx.x;          // 128 threads
    float mu  = colsum[f] * inv_n;
    float var = colsq[f] * inv_n - mu * mu;   // biased variance (torch-style)
    float sc  = gamma[f] * rsqrtf(var + BN_EPS);
    scale[f]  = sc;
    shift[f]  = beta[f] - mu * sc;
}

// --------------------------- fused BN * ReLU --------------------------------
__global__ void k_bn_relu(const float* __restrict__ T,
                          const float* __restrict__ scale,
                          const float* __restrict__ shift,
                          float* __restrict__ Y, long long total)
{
    long long i = ((long long)blockIdx.x * blockDim.x + threadIdx.x) << 2;
    if (i >= total) return;
    float4 t = *(const float4*)(T + i);
    int f = (int)(i & (FDIM - 1));
    float4 r;
    r.x = fmaxf(t.x * scale[f + 0] + shift[f + 0], 0.0f);
    r.y = fmaxf(t.y * scale[f + 1] + shift[f + 1], 0.0f);
    r.z = fmaxf(t.z * scale[f + 2] + shift[f + 2], 0.0f);
    r.w = fmaxf(t.w * scale[f + 3] + shift[f + 3], 0.0f);
    *(float4*)(Y + i) = r;
}

// ------------------------- per-graph mean pool ------------------------------
__global__ void k_pool_accum(const float* __restrict__ X,
                             const long long* __restrict__ batch,
                             float* __restrict__ psum,   // [64,128] pre-zeroed
                             float* __restrict__ pcnt,   // [64]     pre-zeroed
                             long long nnodes)
{
    long long gid  = (long long)blockIdx.x * blockDim.x + threadIdx.x;
    long long node = gid >> 5;
    if (node >= nnodes) return;
    int f4 = (int)(gid & 31) << 2;
    long long g = batch[node];
    float4 v = *(const float4*)(X + node * FDIM + f4);
    float* o = psum + g * FDIM + f4;
    atomicAdd(o + 0, v.x);
    atomicAdd(o + 1, v.y);
    atomicAdd(o + 2, v.z);
    atomicAdd(o + 3, v.w);
    if (f4 == 0) atomicAdd(&pcnt[g], 1.0f);
}

// ----------------------------- classifier -----------------------------------
__global__ void k_logits(const float* __restrict__ psum,
                         const float* __restrict__ pcnt,
                         const float* __restrict__ wc,   // [128,10]
                         const float* __restrict__ bc,   // [10]
                         float* __restrict__ out)        // [64,10]
{
    int t = threadIdx.x;                 // 640 threads, 1 block
    if (t >= NGRAPH * NCLS) return;
    int g = t / NCLS, c = t % NCLS;
    float inv = 1.0f / fmaxf(pcnt[g], 1.0f);
    float acc = bc[c];
    #pragma unroll 4
    for (int f = 0; f < FDIM; ++f)
        acc += psum[g * FDIM + f] * inv * wc[f * NCLS + c];
    out[t] = acc;
}

// ---------------------------------------------------------------------------
extern "C" void kernel_launch(void* const* d_in, const int* in_sizes, int n_in,
                              void* d_out, int out_size, void* d_ws, size_t ws_size,
                              hipStream_t stream)
{
    const long long N = in_sizes[0] / FDIM;      // 50000 nodes
    const long long E = in_sizes[1] / 2;         // 800000 edges
    const long long NF = N * FDIM;

    const float*     X     = (const float*)d_in[0];
    const long long* ei    = (const long long*)d_in[1];
    const long long* srcI  = ei;                 // edge_index[0]
    const long long* dstI  = ei + E;             // edge_index[1]
    const long long* batch = (const long long*)d_in[2];
    // params flatten: 3 layers x (w1,b1,g1,be1,w2,b2,g2,be2), then (wc,bc)
    const float* wc = (const float*)d_in[3 + 3 * 8 + 0];
    const float* bc = (const float*)d_in[3 + 3 * 8 + 1];

    // workspace layout (floats): P | Q | R | stats(512) | pool(64*128+64)
    float* P      = (float*)d_ws;
    float* Q      = P + NF;
    float* R      = Q + NF;
    float* stats  = R + NF;
    float* colsum = stats;          // 128
    float* colsq  = stats + 128;    // 128
    float* scale  = stats + 256;    // 128
    float* shift  = stats + 384;    // 128
    float* psum   = stats + 512;    // 64*128
    float* pcnt   = psum + NGRAPH * FDIM;

    const int ntiles     = (int)((N + 15) / 16);
    const int gemm_grid  = (ntiles + 7) / 8;
    const int scat_grid  = (int)((E * 32 + 255) / 256);
    const int elem_grid  = (int)((NF / 4 + 255) / 256);
    const int pool_grid  = (int)((N * 32 + 255) / 256);
    const float inv_n    = 1.0f / (float)N;

    for (int l = 0; l < 3; ++l) {
        const float* Xcur = (l == 0) ? X : P;
        const float* w1  = (const float*)d_in[3 + l * 8 + 0];
        const float* b1  = (const float*)d_in[3 + l * 8 + 1];
        const float* g1  = (const float*)d_in[3 + l * 8 + 2];
        const float* be1 = (const float*)d_in[3 + l * 8 + 3];
        const float* w2  = (const float*)d_in[3 + l * 8 + 4];
        const float* b2  = (const float*)d_in[3 + l * 8 + 5];
        const float* g2  = (const float*)d_in[3 + l * 8 + 6];
        const float* be2 = (const float*)d_in[3 + l * 8 + 7];

        // h = x + segment_sum(x[src] -> dst)
        hipMemcpyAsync(Q, Xcur, NF * sizeof(float), hipMemcpyDeviceToDevice, stream);
        k_scatter_add<<<scat_grid, 256, 0, stream>>>(Xcur, srcI, dstI, Q, E);

        // t = h @ w1 + b1  (+ BN stats), then BN+ReLU -> Q
        hipMemsetAsync(colsum, 0, 256 * sizeof(float), stream);
        k_gemm_bias_stats<<<gemm_grid, 256, 0, stream>>>(Q, w1, b1, R, colsum, colsq, (int)N);
        k_bn_finalize<<<1, 128, 0, stream>>>(colsum, colsq, g1, be1, scale, shift, inv_n);
        k_bn_relu<<<elem_grid, 256, 0, stream>>>(R, scale, shift, Q, NF);

        // t = y @ w2 + b2  (+ BN stats), then BN+ReLU -> P (next layer input)
        hipMemsetAsync(colsum, 0, 256 * sizeof(float), stream);
        k_gemm_bias_stats<<<gemm_grid, 256, 0, stream>>>(Q, w2, b2, R, colsum, colsq, (int)N);
        k_bn_finalize<<<1, 128, 0, stream>>>(colsum, colsq, g2, be2, scale, shift, inv_n);
        k_bn_relu<<<elem_grid, 256, 0, stream>>>(R, scale, shift, P, NF);
    }

    // global mean pool + classifier
    hipMemsetAsync(psum, 0, (NGRAPH * FDIM + NGRAPH) * sizeof(float), stream);
    k_pool_accum<<<pool_grid, 256, 0, stream>>>(P, batch, psum, pcnt, N);
    k_logits<<<1, 640, 0, stream>>>(psum, pcnt, wc, bc, (float*)d_out);
}
